// PWD_73796128080365
// MI455X (gfx1250) — matrix operand order repaired and checked by reading, not compiled, not run on hardware
//
#include <hip/hip_runtime.h>

typedef __attribute__((ext_vector_type(4)))  float  v4f;
typedef __attribute__((ext_vector_type(8)))  float  v8f;
typedef __attribute__((ext_vector_type(8)))  __bf16 v8bf;
typedef __attribute__((ext_vector_type(16))) __bf16 v16bf;

#define N_PTS   65536
#define D_DIM   64
#define M_CTR   4096
#define TILE_M  16
#define PAIR_M  32                               // 2 tiles staged per buffer
#define NUM_PAIRS (M_CTR / PAIR_M)               // 128
#define WAVES_PER_BLOCK 8
#define PTS_PER_BLOCK   (WAVES_PER_BLOCK * 16)   // 128
#define COL_STRIDE 72   // bf16 elems per centroid column in LDS (64 + 8 pad
                        // -> 144B stride: bank-conflict-free & 16B aligned)

__device__ __forceinline__ void sched_fence() {
#if __has_builtin(__builtin_amdgcn_sched_barrier)
  __builtin_amdgcn_sched_barrier(0);   // nothing may be moved across
#endif
}

// Split an f32 into two bf16 limbs: x ~= hi + lo (16 effective mantissa bits).
__device__ __forceinline__ void bf16_split(float x, __bf16& h, __bf16& l) {
  h = (__bf16)x;
  l = (__bf16)(x - (float)h);
}

__device__ __forceinline__ v8f wmma_bf16(v16bf a, v16bf b, v8f c) {
  return __builtin_amdgcn_wmma_f32_16x16x32_bf16(
      false, a, false, b, (short)0, c, false, false);
}

// Nearest-centroid assignment:
//   dist[n,m] = pn[n] + c_sq[m] - 2*dot(p[n], c[m]);  min+argmin over m.
//
// dot() via bf16x3 limb products on V_WMMA_F32_16X16X32_BF16 (hi*hi + hi*lo
// + lo*hi, f32 accumulate): ~2^-16 relative error, 6 WMMAs per 16x16x64 tile.
// 2 centroid tiles per iteration -> 12 WMMAs in 4 independent chains; all 16
// ds_load_b128 are issued up front (pinned by a scheduling barrier) so DS
// latency is paid once per iteration, not per WMMA.
__global__ __launch_bounds__(256, 1) void PWD_kmeans_assign_wmma_bf16x3(
    const float* __restrict__ p,     // [N, 64]
    const float* __restrict__ pn,    // [N]
    const float* __restrict__ c,     // [M, 64]
    const float* __restrict__ c_sq,  // [M]
    float* __restrict__ out_d,       // [N] min dists
    int*   __restrict__ out_l)       // [N] labels
{
  __shared__ __bf16 sh_hi[2][PAIR_M * COL_STRIDE];   // 2 x 4608 B
  __shared__ __bf16 sh_lo[2][PAIR_M * COL_STRIDE];

  const int tid  = threadIdx.x;
  const int lane = tid & 31;
  const int rit  = lane & 15;   // A: row-in-tile, B: centroid col-in-tile
  const int half = lane >> 4;   // lane half selects K sub-range
  const int wave = tid >> 5;

  const int pbase = blockIdx.x * PTS_PER_BLOCK + wave * 16;

  // ---- Build A limb fragments once (16 points x 64 dims) ----
  // A k order chosen here (same as B), lane holds row (lane&15):
  //   elems 0..15 -> K = 16*half + e
  // k-step s adds 32.
  v16bf a_hi[2], a_lo[2];
  {
    const float* prow = p + (size_t)(pbase + rit) * D_DIM;
    #pragma unroll
    for (int s = 0; s < 2; ++s) {
      const int kb = 32 * s + 16 * half;
      v4f r0 = *(const v4f*)(prow + kb);
      v4f r1 = *(const v4f*)(prow + kb + 4);
      v4f r2 = *(const v4f*)(prow + kb + 8);
      v4f r3 = *(const v4f*)(prow + kb + 12);
      float x[16];
      #pragma unroll
      for (int j = 0; j < 4; ++j) {
        x[j] = r0[j]; x[4 + j] = r1[j]; x[8 + j] = r2[j]; x[12 + j] = r3[j];
      }
      #pragma unroll
      for (int e = 0; e < 16; ++e) {
        __bf16 h, l;
        bf16_split(x[e], h, l);
        a_hi[s][e] = h;
        a_lo[s][e] = l;
      }
    }
  }

  float runmin[8];
  int   runidx[8];
  #pragma unroll
  for (int v = 0; v < 8; ++v) { runmin[v] = __builtin_inff(); runidx[v] = 0; }

  // Cooperative fill mapping: thread -> (column 0..31, 8-float chunk of K).
  const int fcol = tid >> 3;          // 0..31
  const int fkq  = (tid & 7) << 3;    // 0,8,...,56
  const float* fbase = c + (size_t)fcol * D_DIM + fkq;
  const size_t pair_stride = (size_t)PAIR_M * D_DIM;   // floats per 2 tiles

  // ---- Prime buffer 0 with pair 0 ----
  {
    v4f xa = *(const v4f*)(fbase);
    v4f xb = *(const v4f*)(fbase + 4);
    v8bf hv, lv;
    #pragma unroll
    for (int j = 0; j < 4; ++j) {
      __bf16 h, l;
      bf16_split(xa[j], h, l); hv[j] = h;     lv[j] = l;
      bf16_split(xb[j], h, l); hv[4 + j] = h; lv[4 + j] = l;
    }
    *(v8bf*)(&sh_hi[0][fcol * COL_STRIDE + fkq]) = hv;
    *(v8bf*)(&sh_lo[0][fcol * COL_STRIDE + fkq]) = lv;
  }
  __syncthreads();

  for (int pr = 0; pr < NUM_PAIRS; ++pr) {
    const int cur = pr & 1;
    const int nxt = cur ^ 1;
    const int colA = pr * PAIR_M + rit;           // tile A column
    const int colB = colA + TILE_M;               // tile B column
    const float csA = c_sq[colA];
    const float csB = c_sq[colB];

    // Issue next pair's global loads early; latency hides under the WMMAs.
    const bool have_next = (pr + 1 < NUM_PAIRS);
    v4f nxa = {}, nxb = {};
    if (have_next) {
      nxa = *(const v4f*)(fbase + (size_t)(pr + 1) * pair_stride);
      nxb = *(const v4f*)(fbase + (size_t)(pr + 1) * pair_stride + 4);
    }
    __builtin_prefetch(fbase + (size_t)(pr + 2) * pair_stride, 0, 1);

    // ---- B limb fragments from LDS, issued together, first-use order ----
    // B k order chosen here (same as A): lanes 0-15 K=0..15, lanes 16-31 K=16..31,
    // column = lane&15; k-step s adds 32 elements.
    const __bf16* bhA = &sh_hi[cur][rit * COL_STRIDE + half * 16];
    const __bf16* blA = &sh_lo[cur][rit * COL_STRIDE + half * 16];
    const __bf16* bhB = bhA + TILE_M * COL_STRIDE;
    const __bf16* blB = blA + TILE_M * COL_STRIDE;
    v16bf Ah0 = *(const v16bf*)(bhA);
    v16bf Bh0 = *(const v16bf*)(bhB);
    v16bf Al0 = *(const v16bf*)(blA);
    v16bf Bl0 = *(const v16bf*)(blB);
    v16bf Al1 = *(const v16bf*)(blA + 32);
    v16bf Bl1 = *(const v16bf*)(blB + 32);
    v16bf Ah1 = *(const v16bf*)(bhA + 32);
    v16bf Bh1 = *(const v16bf*)(bhB + 32);
    sched_fence();   // keep all ds_loads issued before the WMMA block

    // ---- 12 WMMAs, 4 independent 3-deep accumulation chains ----
    // Per tile: chain0 = hh0 + hl0 + hh1 ; chain1 = lh0 + hl1 + lh1.
    // Consumption order == ds_load issue order -> progressive DS waits.
    v8f c0a = {}, c1a = {}, c0b = {}, c1b = {};
    c0a = wmma_bf16(a_hi[0], Ah0, c0a);
    c0b = wmma_bf16(a_hi[0], Bh0, c0b);
    c1a = wmma_bf16(a_lo[0], Ah0, c1a);
    c1b = wmma_bf16(a_lo[0], Bh0, c1b);
    c0a = wmma_bf16(a_hi[0], Al0, c0a);
    c0b = wmma_bf16(a_hi[0], Bl0, c0b);
    c1a = wmma_bf16(a_hi[1], Al1, c1a);
    c1b = wmma_bf16(a_hi[1], Bl1, c1b);
    c0a = wmma_bf16(a_hi[1], Ah1, c0a);
    c0b = wmma_bf16(a_hi[1], Bh1, c0b);
    c1a = wmma_bf16(a_lo[1], Ah1, c1a);
    c1b = wmma_bf16(a_lo[1], Bh1, c1b);

    // Per-lane running min/argmin. C/D layout: VGPR v holds row (v + half*8),
    // this lane's columns are colA/colB. Strict '<' keeps earliest column;
    // colA < colB and pairs ascend, preserving first-occurrence semantics.
    #pragma unroll
    for (int v = 0; v < 8; ++v) {
      const float dA = c0a[v] + c1a[v];
      const float sA = __builtin_fmaf(-2.0f, dA, csA);
      if (sA < runmin[v]) { runmin[v] = sA; runidx[v] = colA; }
      const float dB = c0b[v] + c1b[v];
      const float sB = __builtin_fmaf(-2.0f, dB, csB);
      if (sB < runmin[v]) { runmin[v] = sB; runidx[v] = colB; }
    }

    // ---- Convert + store next pair into the other LDS buffer ----
    if (have_next) {
      v8bf hv, lv;
      #pragma unroll
      for (int j = 0; j < 4; ++j) {
        __bf16 h, l;
        bf16_split(nxa[j], h, l); hv[j] = h;     lv[j] = l;
        bf16_split(nxb[j], h, l); hv[4 + j] = h; lv[4 + j] = l;
      }
      *(v8bf*)(&sh_hi[nxt][fcol * COL_STRIDE + fkq]) = hv;
      *(v8bf*)(&sh_lo[nxt][fcol * COL_STRIDE + fkq]) = lv;
    }
    __syncthreads();
  }

  // ---- Cross-lane min+argmin over the 16 lanes of each half (wave32) ----
  #pragma unroll
  for (int v = 0; v < 8; ++v) {
    float m  = runmin[v];
    int   id = runidx[v];
    #pragma unroll
    for (int mask = 1; mask < 16; mask <<= 1) {
      const float om = __shfl_xor(m, mask, 32);
      const int   oi = __shfl_xor(id, mask, 32);
      if (om < m || (om == m && oi < id)) { m = om; id = oi; }
    }
    runmin[v] = m;
    runidx[v] = id;
  }

  // Lanes 0 (rows 0..7) and 16 (rows 8..15) write results; pn added here
  // (row-constant, so it does not affect the argmin).
  if (rit == 0) {
    #pragma unroll
    for (int v = 0; v < 8; ++v) {
      const int grow = pbase + half * 8 + v;
      out_d[grow] = runmin[v] + pn[grow];
      out_l[grow] = runidx[v];
    }
  }
}

extern "C" void kernel_launch(void* const* d_in, const int* in_sizes, int n_in,
                              void* d_out, int out_size, void* d_ws, size_t ws_size,
                              hipStream_t stream) {
  (void)in_sizes; (void)n_in; (void)out_size; (void)d_ws; (void)ws_size;

  const float* p    = (const float*)d_in[0];  // [N, 64]
  const float* pn   = (const float*)d_in[1];  // [N, 1]
  const float* c    = (const float*)d_in[2];  // [M, 64]
  const float* c_sq = (const float*)d_in[3];  // [1, M]

  float* out_d = (float*)d_out;               // first N: min dists (f32)
  int*   out_l = (int*)d_out + N_PTS;         // next  N: labels (i32 bits)

  const int blocks = N_PTS / PTS_PER_BLOCK;   // 512
  PWD_kmeans_assign_wmma_bf16x3<<<blocks, 256, 0, stream>>>(p, pn, c, c_sq, out_d, out_l);
}
